// MambaBlock_47665547051930
// MI455X (gfx1250) — compile-verified, matrix operand order checked by reading, not run on hardware
//
#include <hip/hip_runtime.h>

typedef __bf16 bf16;
typedef __attribute__((ext_vector_type(16))) __bf16       v16bf;
typedef __attribute__((ext_vector_type(8)))  float        v8f;
typedef __attribute__((ext_vector_type(8)))  unsigned int v8u;

#define H_DIM   1024
#define DI      2048      // d_inner
#define NS      64        // n_state
#define KC      4         // conv kernel
#define DTR     64        // dt_rank
#define B_SZ    2
#define L_SEQ   2048
#define M_ROWS  (B_SZ * L_SEQ)          // 4096
#define SSM_W   (DTR + 2 * NS)          // 192

// ---------------------------------------------------------------- helpers

static __device__ __forceinline__ bf16 f2bf(float f) {
  unsigned u = __builtin_bit_cast(unsigned, f);
  u += 0x7FFFu + ((u >> 16) & 1u);                  // round-to-nearest-even
  unsigned short s = (unsigned short)(u >> 16);
  return __builtin_bit_cast(bf16, s);
}

static __device__ __forceinline__ float silu(float x) {
  return x * (1.0f / (1.0f + __expf(-x)));
}

// Load one 16x32 bf16 WMMA fragment (A or B). `p` must already point at
//   base + (row_or_col = lane&15)*ld + k0 + (lane>>4)*8
// Per ISA layout: VGPR v<4 -> K = 2v, 2v+1 ; v>=4 -> K = 16+2(v-4) (+8 for
// lanes 16..31, folded into `p`). Pairs are contiguous -> b128-coalescible.
static __device__ __forceinline__ v16bf load_frag(const bf16* __restrict__ p) {
  v8u u;
#pragma unroll
  for (int v = 0; v < 8; ++v) {
    int k = ((v & 3) << 1) + ((v >> 2) << 4);
    u[v] = *(const unsigned int*)(p + k);
  }
  return __builtin_bit_cast(v16bf, u);
}

// Store one 16x16 f32 C/D fragment. Lane layout: col = lane&15,
// rows = (lane>>4)*8 + r for VGPR r in 0..7.
static __device__ __forceinline__ void store_tile(float* __restrict__ C,
                                                  const float* __restrict__ bias,
                                                  v8f c, int row0, int col0,
                                                  int ldc, int lane) {
  int col   = col0 + (lane & 15);
  int rbase = row0 + ((lane >> 4) << 3);
  float bv = bias ? bias[col] : 0.0f;
#pragma unroll
  for (int r = 0; r < 8; ++r)
    C[(size_t)(rbase + r) * ldc + col] = c[r] + bv;
}

// ---------------------------------------------------------------- kernels

__global__ void cvt_f32_bf16(const float* __restrict__ in,
                             bf16* __restrict__ out, long n) {
  long i = (long)blockIdx.x * blockDim.x + threadIdx.x;
  if (i < n) out[i] = f2bf(in[i]);
}

// C[M,N] = A[M,K](bf16,row-major) @ W[N,K](bf16,row-major)^T  (+ bias)
// One wave per 64x32 tile: 4 A-frags x 2 B-frags -> 8 WMMAs per 32-deep
// k-step. Doubles operand reuse vs 32x32 at ~130 VGPRs (wave32).
__global__ void gemm_bf16_wmma(const bf16* __restrict__ A,
                               const bf16* __restrict__ W,
                               const float* __restrict__ bias,
                               float* __restrict__ C,
                               int M, int N, int K) {
  int gtid = blockIdx.x * blockDim.x + threadIdx.x;
  int wave = gtid >> 5;
  int lane = gtid & 31;
  int tilesN = N >> 5;                     // 32-wide col tiles
  int tilesM = M >> 6;                     // 64-tall row tiles
  if (wave >= tilesM * tilesN) return;     // wave-uniform: EXEC stays all-1s
  int tm = (wave / tilesN) << 6;
  int tn = (wave % tilesN) << 5;

  int half = lane >> 4;
  int idx  = lane & 15;
  const bf16* Ab[4];
#pragma unroll
  for (int i = 0; i < 4; ++i)
    Ab[i] = A + (size_t)(tm + 16 * i + idx) * K + half * 8;
  const bf16* Wb0 = W + (size_t)(tn + idx) * K + half * 8;
  const bf16* Wb1 = Wb0 + (size_t)16 * K;

  v8f c[8];
#pragma unroll
  for (int i = 0; i < 8; ++i) c[i] = (v8f){};

  for (int k0 = 0; k0 < K; k0 += 32) {
    v16bf b0 = load_frag(Wb0 + k0);
    v16bf b1 = load_frag(Wb1 + k0);
#pragma unroll
    for (int i = 0; i < 4; ++i) {
      v16bf a = load_frag(Ab[i] + k0);
      c[2 * i]     = __builtin_amdgcn_wmma_f32_16x16x32_bf16(
          false, a, false, b0, (short)0, c[2 * i], false, false);
      c[2 * i + 1] = __builtin_amdgcn_wmma_f32_16x16x32_bf16(
          false, a, false, b1, (short)0, c[2 * i + 1], false, false);
    }
  }
#pragma unroll
  for (int i = 0; i < 4; ++i) {
    store_tile(C, bias, c[2 * i],     tm + 16 * i, tn,      N, lane);
    store_tile(C, bias, c[2 * i + 1], tm + 16 * i, tn + 16, N, lane);
  }
}

// Depthwise causal conv1d + SiLU on x-half of proj; SiLU on res-half.
__global__ void conv_silu_kernel(const float* __restrict__ proj,
                                 const float* __restrict__ conv_w,
                                 const float* __restrict__ conv_b,
                                 float* __restrict__ xconv,
                                 bf16* __restrict__ xconv_bf,
                                 float* __restrict__ res) {
  long i = (long)blockIdx.x * blockDim.x + threadIdx.x;  // over M_ROWS*DI
  if (i >= (long)M_ROWS * DI) return;
  int  d   = (int)(i % DI);
  long row = i / DI;                     // b*L + l
  int  l   = (int)(row % L_SEQ);
  long b0  = row - l;                    // b*L
  float acc = conv_b[d];
#pragma unroll
  for (int j = 0; j < KC; ++j) {
    int ls = l - (KC - 1) + j;
    if (ls >= 0)
      acc += conv_w[d * KC + j] * proj[(size_t)(b0 + ls) * (2 * DI) + d];
  }
  float xv = silu(acc);
  xconv[i]    = xv;
  xconv_bf[i] = f2bf(xv);
  float rv = proj[(size_t)row * (2 * DI) + DI + d];
  res[i] = silu(rv);
}

// Extract dt-columns of ssm to bf16 for the dt GEMM.
__global__ void extract_dt_bf16(const float* __restrict__ ssm,
                                bf16* __restrict__ dtb) {
  int i = blockIdx.x * blockDim.x + threadIdx.x;         // over M_ROWS*DTR
  if (i >= M_ROWS * DTR) return;
  int c = i % DTR, row = i / DTR;
  dtb[i] = f2bf(ssm[(size_t)row * SSM_W + c]);
}

// dt = softplus(dt_raw + 2*b_dt)   (reference adds dt_proj.bias twice)
__global__ void softplus_bias2(float* __restrict__ dt,
                               const float* __restrict__ b_dt) {
  long i = (long)blockIdx.x * blockDim.x + threadIdx.x;  // over M_ROWS*DI
  if (i >= (long)M_ROWS * DI) return;
  int d = (int)(i % DI);
  float v = dt[i] + 2.0f * b_dt[d];
  dt[i] = (v > 20.0f) ? v : log1pf(__expf(v));
}

// Selective scan: one thread per (b,d) channel; h[64] and A-row[64] in VGPRs,
// per-step B/C vectors broadcast through LDS. The 2048-step recurrence is
// latency-bound, so the next step's operands are prefetched
// (global_prefetch_b8) while the current step's exp/FMA chain executes.
__global__ void __launch_bounds__(256, 1)
scan_kernel(const float* __restrict__ xconv, const float* __restrict__ dt,
            const float* __restrict__ ssm, const float* __restrict__ state_in,
            const float* __restrict__ A_log, const float* __restrict__ Dp,
            const float* __restrict__ res, bf16* __restrict__ ybf,
            float* __restrict__ state_out) {
  const int d = blockIdx.x * blockDim.x + threadIdx.x;   // gridDim.x = DI/256
  const int b = blockIdx.y;                              // gridDim.y = B
  __shared__ float Bs[NS];
  __shared__ float Cs[NS];
  float h[NS], Arow[NS];
#pragma unroll
  for (int n = 0; n < NS; ++n) {
    Arow[n] = -__expf(A_log[(size_t)d * NS + n]);
    h[n]    = state_in[((size_t)b * DI + d) * NS + n];
  }
  const float Dv = Dp[d];
  for (int t = 0; t < L_SEQ; ++t) {
    const size_t row = (size_t)b * L_SEQ + t;
    if (threadIdx.x < 2 * NS) {
      float v = ssm[row * SSM_W + DTR + threadIdx.x];
      if (threadIdx.x < NS) Bs[threadIdx.x] = v;
      else                  Cs[threadIdx.x - NS] = v;
    }
    __syncthreads();
    const float xt  = xconv[row * DI + d];
    const float dtt = dt[row * DI + d];
    // Hide next step's global latency under this step's VALU chain.
    if (t + 1 < L_SEQ) {
      const size_t nrow = row + 1;
      __builtin_prefetch(&xconv[nrow * DI + d], 0, 0);
      __builtin_prefetch(&dt[nrow * DI + d], 0, 0);
      __builtin_prefetch(&res[nrow * DI + d], 0, 0);
      if (threadIdx.x < 2 * NS)
        __builtin_prefetch(&ssm[nrow * SSM_W + DTR + threadIdx.x], 0, 0);
    }
    float y = 0.0f;
#pragma unroll
    for (int n = 0; n < NS; ++n) {
      float dA = __expf(dtt * Arow[n]);
      h[n] = fmaf(h[n], dA, (dtt * Bs[n]) * xt);
      y    = fmaf(h[n], Cs[n], y);
    }
    y = fmaf(xt, Dv, y);
    ybf[row * DI + d] = f2bf(y * res[row * DI + d]);
    __syncthreads();
  }
#pragma unroll
  for (int n = 0; n < NS; ++n)
    state_out[((size_t)b * DI + d) * NS + n] = h[n];
}

// ---------------------------------------------------------------- launch

extern "C" void kernel_launch(void* const* d_in, const int* in_sizes, int n_in,
                              void* d_out, int out_size, void* d_ws,
                              size_t ws_size, hipStream_t stream) {
  const float* hs     = (const float*)d_in[0];
  const float* state0 = (const float*)d_in[1];
  const float* W_in   = (const float*)d_in[2];
  const float* b_in   = (const float*)d_in[3];
  const float* conv_w = (const float*)d_in[4];
  const float* conv_b = (const float*)d_in[5];
  const float* W_x    = (const float*)d_in[6];
  const float* W_dt   = (const float*)d_in[7];
  const float* b_dt   = (const float*)d_in[8];
  const float* A_log  = (const float*)d_in[9];
  const float* Dp     = (const float*)d_in[10];
  const float* W_out  = (const float*)d_in[11];
  const float* b_out  = (const float*)d_in[12];

  float* out       = (float*)d_out;                         // [B,L,H]
  float* state_out = out + (size_t)B_SZ * L_SEQ * H_DIM;    // [B,DI,NS]

  // ---- workspace carve-out
  char* w = (char*)d_ws;
  auto alloc = [&](size_t bytes) {
    char* p = w;
    w += (bytes + 255) & ~(size_t)255;
    return p;
  };
  bf16*  hs_bf    = (bf16*)alloc((size_t)M_ROWS * H_DIM * 2);
  bf16*  Win_bf   = (bf16*)alloc((size_t)2 * DI * H_DIM * 2);
  char*  proj_rg  =        alloc((size_t)M_ROWS * 2 * DI * 4);   // 64 MB
  float* proj     = (float*)proj_rg;
  float* xconv_f  = (float*)alloc((size_t)M_ROWS * DI * 4);
  bf16*  xconv_bf = (bf16*) alloc((size_t)M_ROWS * DI * 2);
  float* res_f    = (float*)alloc((size_t)M_ROWS * DI * 4);
  bf16*  Wx_bf    = (bf16*) alloc((size_t)SSM_W * DI * 2);
  float* ssm_f    = (float*)alloc((size_t)M_ROWS * SSM_W * 4);
  bf16*  dtin_bf  = (bf16*) alloc((size_t)M_ROWS * DTR * 2);
  bf16*  Wdt_bf   = (bf16*) alloc((size_t)DI * DTR * 2);
  bf16*  Wout_bf  = (bf16*) alloc((size_t)H_DIM * DI * 2);
  // proj is dead after conv_silu: reuse its space for dt and y*res
  float* dt_f     = (float*)proj_rg;                             // 32 MB
  bf16*  ybf      = (bf16*)(proj_rg + (size_t)M_ROWS * DI * 4);  // 16 MB

  auto blocks_for = [](long n) { return (int)((n + 255) / 256); };
  auto gemm = [&](const bf16* A, const bf16* Wm, const float* bias, float* C,
                  int M, int N, int K) {
    long threads = (long)(M >> 6) * (N >> 5) * 32;   // one wave per 64x32 tile
    gemm_bf16_wmma<<<blocks_for(threads), 256, 0, stream>>>(A, Wm, bias, C,
                                                            M, N, K);
  };

  // 1) in_proj
  cvt_f32_bf16<<<blocks_for((long)M_ROWS * H_DIM), 256, 0, stream>>>(
      hs, hs_bf, (long)M_ROWS * H_DIM);
  cvt_f32_bf16<<<blocks_for((long)2 * DI * H_DIM), 256, 0, stream>>>(
      W_in, Win_bf, (long)2 * DI * H_DIM);
  gemm(hs_bf, Win_bf, b_in, proj, M_ROWS, 2 * DI, H_DIM);

  // 2) depthwise causal conv + SiLU (x) / SiLU (res)
  conv_silu_kernel<<<blocks_for((long)M_ROWS * DI), 256, 0, stream>>>(
      proj, conv_w, conv_b, xconv_f, xconv_bf, res_f);

  // 3) x_proj -> [dt | B | C]
  cvt_f32_bf16<<<blocks_for((long)SSM_W * DI), 256, 0, stream>>>(
      W_x, Wx_bf, (long)SSM_W * DI);
  gemm(xconv_bf, Wx_bf, nullptr, ssm_f, M_ROWS, SSM_W, DI);

  // 4) dt_proj + softplus (bias applied twice, faithful to reference)
  extract_dt_bf16<<<blocks_for((long)M_ROWS * DTR), 256, 0, stream>>>(
      ssm_f, dtin_bf);
  cvt_f32_bf16<<<blocks_for((long)DI * DTR), 256, 0, stream>>>(
      W_dt, Wdt_bf, (long)DI * DTR);
  gemm(dtin_bf, Wdt_bf, nullptr, dt_f, M_ROWS, DI, DTR);
  softplus_bias2<<<blocks_for((long)M_ROWS * DI), 256, 0, stream>>>(dt_f, b_dt);

  // 5) selective scan -> y*res (bf16) + final state (f32, output #2)
  {
    dim3 grid(DI / 256, B_SZ);
    scan_kernel<<<grid, 256, 0, stream>>>(xconv_f, dt_f, ssm_f, state0, A_log,
                                          Dp, res_f, ybf, state_out);
  }

  // 6) out_proj -> output #1
  cvt_f32_bf16<<<blocks_for((long)H_DIM * DI), 256, 0, stream>>>(
      W_out, Wout_bf, (long)H_DIM * DI);
  gemm(ybf, Wout_bf, b_out, out, M_ROWS, H_DIM, DI);
}
